// ShapeEmbedding_21655225106935
// MI455X (gfx1250) — matrix operand order repaired and checked by reading, not compiled, not run on hardware
//
#include <hip/hip_runtime.h>
#include <hip/hip_bf16.h>

typedef __attribute__((ext_vector_type(16))) __bf16 v16bf;
typedef __attribute__((ext_vector_type(8)))  float  v8f;

union Frag16 { unsigned short u[16]; v16bf v; };

__device__ __forceinline__ unsigned short f2bf(float f) {
    unsigned int u = __float_as_uint(f);
    u += 0x7FFFu + ((u >> 16) & 1u);   // round-to-nearest-even
    return (unsigned short)(u >> 16);
}

// ---------------- elementwise / degree kernels ----------------

__global__ void k_fill(float* __restrict__ p, float v, int n) {
    int i = blockIdx.x * blockDim.x + threadIdx.x;
    if (i < n) p[i] = v;
}

__global__ void k_deg(const long long* __restrict__ col, float* __restrict__ deg, int E) {
    int e = blockIdx.x * blockDim.x + threadIdx.x;
    if (e < E) atomicAdd(&deg[(int)col[e]], 1.0f);
}

__global__ void k_rsqrt(float* __restrict__ p, int n) {
    int i = blockIdx.x * blockDim.x + threadIdx.x;
    if (i < n) p[i] = rsqrtf(p[i]);
}

__global__ void k_leaky(float* __restrict__ p, int n) {
    int i = blockIdx.x * blockDim.x + threadIdx.x;
    if (i < n) { float x = p[i]; p[i] = x >= 0.0f ? x : 0.01f * x; }
}

// ---------------- WMMA GEMM: Y[N x OUT] = act(X[N x IN] @ W[IN x OUT] + bias) --------
// 32 nodes per block, 8 waves; each wave owns 16x16 output tiles, K-loop of 16x16x32 bf16 WMMAs.
// A staged row-major (fragment = two contiguous 16B runs -> 2x ds_load_b128).
// W staged pre-swizzled into B-fragment layout (fragment = one contiguous 32B run -> 2x ds_load_b128).

template<int IN, int OUT>
__global__ __launch_bounds__(256) void k_gemm(const float* __restrict__ X,
                                              const float* __restrict__ W,
                                              const float* __restrict__ bias,
                                              float* __restrict__ Y, int act) {
    constexpr int NT = OUT / 16;       // column tiles
    constexpr int NK = IN / 32;        // K steps
    __shared__ unsigned short Alds[32 * IN];
    __shared__ unsigned short Bswz[IN * OUT];
    const int tid = threadIdx.x;
    const int nb  = blockIdx.x * 32;

    // Stage W swizzled: element i -> (tile = i>>9, lane = (i>>4)&31, j = i&15)
    // tile = kk*NT + nt ; value = W[(kk*32 + (lane>=16?8:0) + (j&7) + (j>=8?16:0)) * OUT + nt*16 + (lane&15)]
    for (int i = tid; i < IN * OUT; i += 256) {
        int j    = i & 15;
        int ln   = (i >> 4) & 31;
        int tile = i >> 9;
        int nt   = tile % NT;
        int kk   = tile / NT;
        int k    = kk * 32 + (ln >= 16 ? 8 : 0) + (j & 7) + (j >= 8 ? 16 : 0);
        int n    = nt * 16 + (ln & 15);
        Bswz[i]  = f2bf(W[k * OUT + n]);
    }
    // Stage A row-major (bf16)
    for (int i = tid; i < 32 * IN; i += 256) {
        int r = i / IN, c = i % IN;
        Alds[i] = f2bf(X[(size_t)(nb + r) * IN + c]);
    }
    __syncthreads();

    const int  wave  = tid >> 5;
    const int  lane  = tid & 31;
    const int  nhalf = lane & 15;
    const bool hi    = lane >= 16;
    for (int t = wave; t < 2 * NT; t += 8) {
        const int mt = t / NT, nt = t % NT;
        v8f acc = {};
        #pragma unroll
        for (int kk = 0; kk < NK; ++kk) {
            Frag16 a, b;
            // A fragment (16x32 bf16): lanes 0-15 K 0..7 & 16..23; lanes 16-31 K 8..15 & 24..31
            const int m  = mt * 16 + nhalf;
            const int ka = kk * 32 + (hi ? 8 : 0);
            #pragma unroll
            for (int j = 0; j < 8; ++j) {
                a.u[j]     = Alds[m * IN + ka + j];
                a.u[j + 8] = Alds[m * IN + ka + 16 + j];
            }
            // B fragment: contiguous 16 ushorts in swizzled LDS
            const int base = ((kk * NT + nt) * 32 + lane) * 16;
            #pragma unroll
            for (int j = 0; j < 16; ++j)
                b.u[j] = Bswz[base + j];
            acc = __builtin_amdgcn_wmma_f32_16x16x32_bf16(
                    false, a.v, false, b.v, (short)0, acc, false, false);
        }
        // D tile: VGPR r -> M = r (lanes 0-15) / r+8 (lanes 16-31), N = lane&15
        const int  nc = nt * 16 + nhalf;
        const float bv = bias ? bias[nc] : 0.0f;
        #pragma unroll
        for (int r = 0; r < 8; ++r) {
            int m = mt * 16 + r + (hi ? 8 : 0);
            float f = acc[r] + bv;
            if (act) f = f >= 0.0f ? f : 0.01f * f;
            Y[(size_t)(nb + m) * OUT + nc] = f;
        }
    }
}

// ---------------- GCN aggregation ----------------

// agg[i][c] = dis[i]^2 * h[i][c] + bias[c]   (self-loop + bias)
template<int OUT>
__global__ void k_selfloop(const float* __restrict__ h, const float* __restrict__ dis,
                           const float* __restrict__ bias, float* __restrict__ agg, int n) {
    int i = blockIdx.x * blockDim.x + threadIdx.x;
    if (i >= n * OUT) return;
    int node = i / OUT, c = i % OUT;
    float d = dis[node];
    agg[i] = d * d * h[i] + bias[c];
}

// one wave per edge: agg[col] += h[row] * (dis[row]*dis[col])
template<int OUT>
__global__ __launch_bounds__(256) void k_edge(const long long* __restrict__ rowI,
                                              const long long* __restrict__ colI,
                                              const float* __restrict__ h,
                                              const float* __restrict__ dis,
                                              float* __restrict__ agg, int E) {
    constexpr int CPL = OUT / 32;   // channels per lane
    const int wid  = (blockIdx.x * 256 + threadIdx.x) >> 5;
    const int lane = threadIdx.x & 31;
    if (wid >= E) return;
    const int r = (int)rowI[wid];
    const int c = (int)colI[wid];
    const float norm = dis[r] * dis[c];
    const float* hr = h   + (size_t)r * OUT + lane * CPL;
    float*       ac = agg + (size_t)c * OUT + lane * CPL;
    float v[CPL];
    #pragma unroll
    for (int i = 0; i < CPL; ++i) v[i] = hr[i] * norm;
    #pragma unroll
    for (int i = 0; i < CPL; ++i) atomicAdd(&ac[i], v[i]);
}

// ---------------- mean over nodes (64 channels) ----------------

__global__ void k_mean(const float* __restrict__ x, float* __restrict__ out, int n) {
    int c    = threadIdx.x & 63;
    int rgrp = threadIdx.x >> 6;                 // 0..3
    int stride = gridDim.x * 4;
    float s = 0.0f;
    for (int r = blockIdx.x * 4 + rgrp; r < n; r += stride)
        s += x[(size_t)r * 64 + c];
    atomicAdd(&out[c], s);
}

__global__ void k_scale(float* __restrict__ out, float s, int n) {
    int i = threadIdx.x;
    if (i < n) out[i] *= s;
}

// ---------------- launcher ----------------

extern "C" void kernel_launch(void* const* d_in, const int* in_sizes, int n_in,
                              void* d_out, int out_size, void* d_ws, size_t ws_size,
                              hipStream_t stream) {
    const float* pose  = (const float*)d_in[0];
    const float* w_pos = (const float*)d_in[1];
    const float* b_pos = (const float*)d_in[2];
    const float* w_fc  = (const float*)d_in[3];
    const float* b_fc  = (const float*)d_in[4];
    const float* w_g1  = (const float*)d_in[5];
    const float* b_g1  = (const float*)d_in[6];
    const float* w_g2  = (const float*)d_in[7];
    const float* b_g2  = (const float*)d_in[8];
    const float* w_g3  = (const float*)d_in[9];
    const float* b_g3  = (const float*)d_in[10];
    const long long* ei = (const long long*)d_in[11];

    const int N = in_sizes[0] / 64;        // 100000 (divisible by 32)
    const int E = in_sizes[11] / 2;        // 1600000
    const long long* row = ei;
    const long long* col = ei + E;
    float* out = (float*)d_out;

    float* dis = (float*)d_ws;
    size_t off = (((size_t)N * 4) + 511) & ~(size_t)511;
    float* bufA = (float*)((char*)d_ws + off);
    float* bufB = bufA + (size_t)N * 128;
    float* bufC = bufB + (size_t)N * 128;

    const int TPB = 256;
    const int gN   = (N + TPB - 1) / TPB;
    const int gE   = (E + TPB - 1) / TPB;
    const int gEw  = (E + 7) / 8;                 // 8 waves (edges) per block
    const int g128 = (N * 128 + TPB - 1) / TPB;
    const int g64  = (N * 64 + TPB - 1) / TPB;
    const int gG   = N / 32;                      // GEMM blocks

    // degrees with self-loops -> deg_inv_sqrt (in place)
    k_fill<<<gN, TPB, 0, stream>>>(dis, 1.0f, N);
    k_deg<<<gE, TPB, 0, stream>>>(col, dis, E);
    k_rsqrt<<<gN, TPB, 0, stream>>>(dis, N);

    // RefineNetwork: leaky(pose@Wpos+b) @ Wfc + b  -> bufA
    k_gemm<64, 128><<<gG, TPB, 0, stream>>>(pose, w_pos, b_pos, bufB, 1);
    k_gemm<128, 128><<<gG, TPB, 0, stream>>>(bufB, w_fc, b_fc, bufA, 0);

    // conv1: x=bufA -> h=bufB -> agg=bufC
    k_gemm<128, 128><<<gG, TPB, 0, stream>>>(bufA, w_g1, nullptr, bufB, 0);
    k_selfloop<128><<<g128, TPB, 0, stream>>>(bufB, dis, b_g1, bufC, N);
    k_edge<128><<<gEw, TPB, 0, stream>>>(row, col, bufB, dis, bufC, E);
    k_leaky<<<g128, TPB, 0, stream>>>(bufC, N * 128);

    // conv2: x=bufC -> h=bufA -> agg=bufB
    k_gemm<128, 128><<<gG, TPB, 0, stream>>>(bufC, w_g2, nullptr, bufA, 0);
    k_selfloop<128><<<g128, TPB, 0, stream>>>(bufA, dis, b_g2, bufB, N);
    k_edge<128><<<gEw, TPB, 0, stream>>>(row, col, bufA, dis, bufB, E);
    k_leaky<<<g128, TPB, 0, stream>>>(bufB, N * 128);

    // conv3 (128->64): x=bufB -> h=bufC -> agg=bufA
    k_gemm<128, 64><<<gG, TPB, 0, stream>>>(bufB, w_g3, nullptr, bufC, 0);
    k_selfloop<64><<<g64, TPB, 0, stream>>>(bufC, dis, b_g3, bufA, N);
    k_edge<64><<<gEw, TPB, 0, stream>>>(row, col, bufC, dis, bufA, E);
    k_leaky<<<g64, TPB, 0, stream>>>(bufA, N * 64);

    // mean over nodes
    k_fill<<<1, TPB, 0, stream>>>(out, 0.0f, 64);
    k_mean<<<256, TPB, 0, stream>>>(bufA, out, N);
    k_scale<<<1, 64, 0, stream>>>(out, 1.0f / (float)N, 64);
}
